// MultiheadAttender_74388833567363
// MI455X (gfx1250) — matrix operand order repaired and checked by reading, not compiled, not run on hardware
//
#include <hip/hip_runtime.h>

typedef unsigned short u16t;
typedef __attribute__((ext_vector_type(16))) __bf16 v16bf;
typedef __attribute__((ext_vector_type(8)))  float  v8f;
typedef __attribute__((ext_vector_type(4)))  unsigned int u32x4;

union Frag { v16bf v; u32x4 u[2]; };

__device__ __forceinline__ u16t f2bf(float f) {
  unsigned int u = __float_as_uint(f);
  u += 0x7FFFu + ((u >> 16) & 1u);   // round-to-nearest-even
  return (u16t)(u >> 16);
}

// ---------------------------------------------------------------------------
// Elementwise f32 -> bf16
__global__ __launch_bounds__(256) void cvt_bf16(const float* __restrict__ in,
                                                u16t* __restrict__ out, int n) {
  int i = blockIdx.x * 256 + threadIdx.x;
  if (i < n) out[i] = f2bf(in[i]);
}

// Transpose (rows x cols) f32 -> bf16 (cols x rows)
__global__ __launch_bounds__(256) void cvt_bf16_t(const float* __restrict__ in,
                                                  u16t* __restrict__ out,
                                                  int rows, int cols) {
  int i = blockIdx.x * 256 + threadIdx.x;
  if (i < rows * cols) {
    int r = i / cols, c = i % cols;
    out[(size_t)c * rows + r] = f2bf(in[i]);
  }
}

// ---------------------------------------------------------------------------
// Projection GEMM:  Y(8192 x 2048) = A(8192 x 256, bf16) * Bt^T
// Bt stored transposed: [2048][256] bf16 (k-contiguous per output column).
// Register-blocked: each wave owns a 64x64 super-tile (4x4 WMMA tiles),
// 16 WMMAs per 16 b128 loads per k-step.
// mode 0: out bf16 [b][h][m][d]   (Q / K head-major)
// mode 1: out bf16 [b][h][d][n]   (V transposed per head)
__global__ __launch_bounds__(256) void proj_gemm(const u16t* __restrict__ A,
                                                 const u16t* __restrict__ Bt,
                                                 u16t* __restrict__ Out,
                                                 int mode) {
  const int wave = threadIdx.x >> 5, lane = threadIdx.x & 31;
  const int half = lane >> 4, l16 = lane & 15;
  const int wt = blockIdx.x * 8 + wave;     // 4096 super-tiles
  const int mt = wt & 127;                  // 128 M super-tiles (64 rows)
  const int nt = wt >> 7;                   // 32  N super-tiles (64 cols)

  const u16t* arow[4];
  const u16t* brow[4];
#pragma unroll
  for (int i = 0; i < 4; ++i)
    arow[i] = A + (size_t)(mt * 64 + i * 16 + l16) * 256;
#pragma unroll
  for (int j = 0; j < 4; ++j)
    brow[j] = Bt + (size_t)(nt * 64 + j * 16 + l16) * 256;

  v8f c[4][4];
#pragma unroll
  for (int i = 0; i < 4; ++i)
#pragma unroll
    for (int j = 0; j < 4; ++j) { v8f z = {}; c[i][j] = z; }

#pragma unroll
  for (int t = 0; t < 8; ++t) {
    Frag a[4], b[4];
#pragma unroll
    for (int i = 0; i < 4; ++i) {
      a[i].u[0] = *(const u32x4*)(arow[i] + t * 32 + 8 * half);
      a[i].u[1] = *(const u32x4*)(arow[i] + t * 32 + 16 + 8 * half);
    }
#pragma unroll
    for (int j = 0; j < 4; ++j) {
      b[j].u[0] = *(const u32x4*)(brow[j] + t * 32 + 8 * half);
      b[j].u[1] = *(const u32x4*)(brow[j] + t * 32 + 16 + 8 * half);
    }
#pragma unroll
    for (int i = 0; i < 4; ++i)
#pragma unroll
      for (int j = 0; j < 4; ++j)
        c[i][j] = __builtin_amdgcn_wmma_f32_16x16x32_bf16(
            false, a[i].v, false, b[j].v, (short)0, c[i][j], false, false);
  }

#pragma unroll
  for (int j = 0; j < 4; ++j) {
    const int ng = nt * 64 + j * 16 + l16;
    const int h = ng >> 8, d = ng & 255;
#pragma unroll
    for (int i = 0; i < 4; ++i) {
#pragma unroll
      for (int r = 0; r < 8; ++r) {
        int rg = mt * 64 + i * 16 + r + 8 * half;   // global row (b*2048 + m)
        int b_ = rg >> 11, m = rg & 2047;
        size_t idx = (mode == 0)
            ? ((((size_t)(b_ * 8 + h)) * 2048 + m) * 256 + d)
            : ((((size_t)(b_ * 8 + h)) * 256 + d) * 2048 + m);
        Out[idx] = f2bf(c[i][j][r]);
      }
    }
  }
}

// ---------------------------------------------------------------------------
// Flash attention per (b,h): S = Q K^T * scale, online softmax, O = P V.
// Qp: [bh][m][d] bf16, Kp: [bh][n][d] bf16, Vp: [bh][d][n] bf16.
// Ob out: [b][m][h][d] bf16 (== rows of the final GEMM, k = h*256+d).
__global__ __launch_bounds__(256) void attn_kernel(const u16t* __restrict__ Qp,
                                                   const u16t* __restrict__ Kp,
                                                   const u16t* __restrict__ Vp,
                                                   u16t* __restrict__ Ob) {
  __shared__ u16t kT[32 * 256];     // 16 KB: K rows n0..n0+31
  __shared__ u16t vT[256 * 32];     // 16 KB: V^T cols n0..n0+31 for all d
  __shared__ u16t pS[8][16 * 32];   // per-wave P transpose scratch

  const int bh   = blockIdx.x >> 4;            // 0..31
  const int mblk = blockIdx.x & 15;
  const int wave = threadIdx.x >> 5;
  const int lane = threadIdx.x & 31;
  const int half = lane >> 4, l16 = lane & 15;
  const int m0   = mblk * 128 + wave * 16;

  const u16t* Qbase = Qp + (size_t)bh * 2048 * 256;
  const u16t* Kbase = Kp + (size_t)bh * 2048 * 256;
  const u16t* Vbase = Vp + (size_t)bh * 256 * 2048;

  // Q fragments for all 8 k-tiles (held in VGPRs for the whole N loop)
  Frag qf[8];
  {
    const u16t* qrow = Qbase + (size_t)(m0 + l16) * 256;
#pragma unroll
    for (int t = 0; t < 8; ++t) {
      qf[t].u[0] = *(const u32x4*)(qrow + t * 32 + 8 * half);
      qf[t].u[1] = *(const u32x4*)(qrow + t * 32 + 16 + 8 * half);
    }
  }

  v8f o[16];
#pragma unroll
  for (int c = 0; c < 16; ++c) { v8f z = {}; o[c] = z; }
  float Mrow[8], Lrow[8];
#pragma unroll
  for (int r = 0; r < 8; ++r) { Mrow[r] = -1e30f; Lrow[r] = 0.f; }

  const float scale = 0.0625f;   // 1/sqrt(256)

  for (int n0 = 0; n0 < 2048; n0 += 32) {
    __syncthreads();             // previous-iteration LDS reads complete
    {
      // K rows n0..n0+31 are contiguous: straight 16 KB copy
      const u32x4* ksrc = (const u32x4*)(Kbase + (size_t)n0 * 256);
      u32x4* kdst = (u32x4*)kT;
#pragma unroll
      for (int it = 0; it < 4; ++it) {
        int i = threadIdx.x + 256 * it;
        kdst[i] = ksrc[i];
      }
      // V^T tile: 32 ushorts from each of the 256 d-rows
      u32x4* vdst = (u32x4*)vT;
#pragma unroll
      for (int it = 0; it < 4; ++it) {
        int i = threadIdx.x + 256 * it;
        int d = i >> 2, ch = i & 3;
        vdst[i] = *(const u32x4*)(Vbase + (size_t)d * 2048 + n0 + ch * 8);
      }
    }
    __syncthreads();

    // S tiles: 16x16 x2 (cols n0..n0+15 and n0+16..n0+31)
    v8f s0 = {}, s1 = {};
#pragma unroll
    for (int t = 0; t < 8; ++t) {
      Frag kf;
      const u16t* kr0 = kT + (size_t)l16 * 256 + t * 32 + 8 * half;
      kf.u[0] = *(const u32x4*)(kr0);
      kf.u[1] = *(const u32x4*)(kr0 + 16);
      s0 = __builtin_amdgcn_wmma_f32_16x16x32_bf16(false, qf[t].v, false, kf.v,
                                                   (short)0, s0, false, false);
      const u16t* kr1 = kT + (size_t)(16 + l16) * 256 + t * 32 + 8 * half;
      kf.u[0] = *(const u32x4*)(kr1);
      kf.u[1] = *(const u32x4*)(kr1 + 16);
      s1 = __builtin_amdgcn_wmma_f32_16x16x32_bf16(false, qf[t].v, false, kf.v,
                                                   (short)0, s1, false, false);
    }

    // Online softmax. Row m = r + 8*half lives in one 16-lane group.
    float f[8];
#pragma unroll
    for (int r = 0; r < 8; ++r) {
      float a0 = s0[r] * scale;
      float a1 = s1[r] * scale;
      float mx = fmaxf(a0, a1);
#pragma unroll
      for (int msk = 8; msk >= 1; msk >>= 1)
        mx = fmaxf(mx, __shfl_xor(mx, msk, 16));
      float nm = fmaxf(Mrow[r], mx);
      float p0 = __expf(a0 - nm);
      float p1 = __expf(a1 - nm);
      f[r] = __expf(Mrow[r] - nm);
      Mrow[r] = nm;
      float ls = p0 + p1;
#pragma unroll
      for (int msk = 8; msk >= 1; msk >>= 1)
        ls += __shfl_xor(ls, msk, 16);
      Lrow[r] = Lrow[r] * f[r] + ls;
      int m = r + 8 * half;
      pS[wave][m * 32 + l16]      = f2bf(p0);
      pS[wave][m * 32 + 16 + l16] = f2bf(p1);
    }

    // same-wave DS RAW fence before re-reading P in A-layout
    asm volatile("s_wait_dscnt 0" ::: "memory");

    Frag pf;
    {
      const u16t* pr = &pS[wave][l16 * 32];
      pf.u[0] = *(const u32x4*)(pr + 8 * half);
      pf.u[1] = *(const u32x4*)(pr + 16 + 8 * half);
    }

    // O update across all 16 d_v tiles (rescale accumulators first)
#pragma unroll
    for (int c = 0; c < 16; ++c) {
#pragma unroll
      for (int r = 0; r < 8; ++r) o[c][r] = o[c][r] * f[r];
      Frag vf;
      const u16t* vr = vT + (size_t)(c * 16 + l16) * 32 + 8 * half;
      vf.u[0] = *(const u32x4*)(vr);
      vf.u[1] = *(const u32x4*)(vr + 16);
      o[c] = __builtin_amdgcn_wmma_f32_16x16x32_bf16(false, pf.v, false, vf.v,
                                                     (short)0, o[c], false, false);
    }
  }

  // Epilogue: normalize and store bf16 to [b][m][h][d]
  const int b_ = bh >> 3, h = bh & 7;
  float inv[8];
#pragma unroll
  for (int r = 0; r < 8; ++r) inv[r] = 1.0f / Lrow[r];
#pragma unroll
  for (int c = 0; c < 16; ++c) {
#pragma unroll
    for (int r = 0; r < 8; ++r) {
      int m = m0 + r + 8 * half;
      int d = c * 16 + l16;
      Ob[(((size_t)b_ * 2048 + m) * 8 + h) * 256 + d] = f2bf(o[c][r] * inv[r]);
    }
  }
}

// ---------------------------------------------------------------------------
// Output GEMM: out(8192 x 256, f32) = Ob(8192 x 2048 bf16) * Wu + bu
// WuT stored transposed: [256][2048] bf16.
// Register-blocked 32x32 per wave (2x2 WMMA tiles) to keep 2048 waves alive.
__global__ __launch_bounds__(256) void final_gemm(const u16t* __restrict__ A,
                                                  const u16t* __restrict__ Bt,
                                                  const float* __restrict__ bias,
                                                  float* __restrict__ Out) {
  const int wave = threadIdx.x >> 5, lane = threadIdx.x & 31;
  const int half = lane >> 4, l16 = lane & 15;
  const int wt = blockIdx.x * 8 + wave;     // 2048 super-tiles
  const int mt = wt & 255;                  // 256 M super-tiles (32 rows)
  const int nt = wt >> 8;                   // 8   N super-tiles (32 cols)

  const u16t* arow[2];
  const u16t* brow[2];
#pragma unroll
  for (int i = 0; i < 2; ++i)
    arow[i] = A + (size_t)(mt * 32 + i * 16 + l16) * 2048;
#pragma unroll
  for (int j = 0; j < 2; ++j)
    brow[j] = Bt + (size_t)(nt * 32 + j * 16 + l16) * 2048;

  v8f c[2][2];
#pragma unroll
  for (int i = 0; i < 2; ++i)
#pragma unroll
    for (int j = 0; j < 2; ++j) { v8f z = {}; c[i][j] = z; }

#pragma unroll 4
  for (int t = 0; t < 64; ++t) {
    Frag a[2], b[2];
#pragma unroll
    for (int i = 0; i < 2; ++i) {
      a[i].u[0] = *(const u32x4*)(arow[i] + t * 32 + 8 * half);
      a[i].u[1] = *(const u32x4*)(arow[i] + t * 32 + 16 + 8 * half);
    }
#pragma unroll
    for (int j = 0; j < 2; ++j) {
      b[j].u[0] = *(const u32x4*)(brow[j] + t * 32 + 8 * half);
      b[j].u[1] = *(const u32x4*)(brow[j] + t * 32 + 16 + 8 * half);
    }
#pragma unroll
    for (int i = 0; i < 2; ++i)
#pragma unroll
      for (int j = 0; j < 2; ++j)
        c[i][j] = __builtin_amdgcn_wmma_f32_16x16x32_bf16(
            false, a[i].v, false, b[j].v, (short)0, c[i][j], false, false);
  }

#pragma unroll
  for (int j = 0; j < 2; ++j) {
    const int ng = nt * 32 + j * 16 + l16;
    const float bv = bias[ng];
#pragma unroll
    for (int i = 0; i < 2; ++i) {
#pragma unroll
      for (int r = 0; r < 8; ++r) {
        int rg = mt * 32 + i * 16 + r + 8 * half;
        Out[(size_t)rg * 256 + ng] = c[i][j][r] + bv;
      }
    }
  }
}

// ---------------------------------------------------------------------------
extern "C" void kernel_launch(void* const* d_in, const int* in_sizes, int n_in,
                              void* d_out, int out_size, void* d_ws, size_t ws_size,
                              hipStream_t stream) {
  const float* q  = (const float*)d_in[0];
  const float* k  = (const float*)d_in[1];
  const float* v  = (const float*)d_in[2];
  const float* Wq = (const float*)d_in[3];
  const float* Wk = (const float*)d_in[4];
  const float* Wv = (const float*)d_in[5];
  const float* Wu = (const float*)d_in[6];
  const float* bu = (const float*)d_in[7];
  float* out = (float*)d_out;

  u16t* ws = (u16t*)d_ws;
  const size_t NXD = 8192u * 256u;          // 2,097,152
  const size_t NW  = 2048u * 256u;          // 524,288
  const size_t NP  = 32u * 2048u * 256u;    // 16,777,216
  u16t* qb   = ws;
  u16t* kb   = qb + NXD;
  u16t* vb   = kb + NXD;
  u16t* WqT  = vb + NXD;
  u16t* WkT  = WqT + NW;
  u16t* WvT  = WkT + NW;
  u16t* WuT  = WvT + NW;
  u16t* Qp   = WuT + NW;
  u16t* Kp   = Qp + NP;
  u16t* Vp   = Kp + NP;
  u16t* Obuf = Vp + NP;                     // 8192 x 2048

  // Stage 1: conversions
  cvt_bf16<<<(int)(NXD / 256), 256, 0, stream>>>(q, qb, (int)NXD);
  cvt_bf16<<<(int)(NXD / 256), 256, 0, stream>>>(k, kb, (int)NXD);
  cvt_bf16<<<(int)(NXD / 256), 256, 0, stream>>>(v, vb, (int)NXD);
  cvt_bf16_t<<<(int)(NW / 256), 256, 0, stream>>>(Wq, WqT, 256, 2048);
  cvt_bf16_t<<<(int)(NW / 256), 256, 0, stream>>>(Wk, WkT, 256, 2048);
  cvt_bf16_t<<<(int)(NW / 256), 256, 0, stream>>>(Wv, WvT, 256, 2048);
  cvt_bf16_t<<<(int)(NW / 256), 256, 0, stream>>>(Wu, WuT, 2048, 256);

  // Stage 2: head projections (4096 64x64 super-tiles, 8 waves/block)
  proj_gemm<<<512, 256, 0, stream>>>(qb, WqT, Qp, 0);
  proj_gemm<<<512, 256, 0, stream>>>(kb, WkT, Kp, 0);
  proj_gemm<<<512, 256, 0, stream>>>(vb, WvT, Vp, 1);

  // Stage 3: flash attention — 32 (b,h) pairs x 16 row-blocks of 128
  attn_kernel<<<512, 256, 0, stream>>>(Qp, Kp, Vp, Obuf);

  // Stage 4: output projection + bias (2048 32x32 super-tiles)
  final_gemm<<<256, 256, 0, stream>>>(Obuf, WuT, bu, out);
}
